// AttentionBlock_1786706395712
// MI455X (gfx1250) — matrix operand order repaired and checked by reading, not compiled, not run on hardware
//
#include <hip/hip_runtime.h>

typedef __attribute__((ext_vector_type(16))) _Float16 v16h;
typedef __attribute__((ext_vector_type(8)))  _Float16 v8h;
typedef __attribute__((ext_vector_type(8)))  float    v8f;

#define WMMA_F16(a, b, c) \
  __builtin_amdgcn_wmma_f32_16x16x32_f16(false, (a), false, (b), (short)0, (c), false, false)

// Problem dims
constexpr int Bn  = 8;     // batch
constexpr int Cch = 512;   // channels
constexpr int Np  = 1024;  // tokens (32*32)
constexpr int Hh  = 8;     // heads
constexpr int Dd  = 64;    // head dim

__device__ __forceinline__ v16h cat8(v8h lo, v8h hi8) {
  v16h r;
#pragma unroll
  for (int i = 0; i < 8; ++i) { r[i] = lo[i]; r[8 + i] = hi8[i]; }
  return r;
}

// LDS aperture: flat shared address low 32 bits are the LDS byte address.
__device__ __forceinline__ unsigned lds_off(const void* p) {
  return (unsigned)(size_t)p;
}

// CDNA5 async global->LDS copy (per-lane LDS scatter), tracked by ASYNCcnt.
__device__ __forceinline__ void async_g2l_b32(unsigned lds, const float* g) {
  asm volatile("global_load_async_to_lds_b32 %0, %1, off"
               :: "v"(lds), "v"(g) : "memory");
}

// ---------------------------------------------------------------------------
// Kernel 0: generic 32x32-tiled transpose + f32->f16 convert.
//   src: [batch][K][J] f32   ->   dst: [batch][J][K] f16
// Global reads land *transposed* in LDS via async-to-LDS per-lane scatter.
// ---------------------------------------------------------------------------
__global__ __launch_bounds__(256) void transpose_cvt_kernel(
    const float* __restrict__ src, _Float16* __restrict__ dst, int K, int J) {
  __shared__ __align__(16) float tile[32 * 33];  // [j_local][k_local], padded
  const size_t bofs = (size_t)blockIdx.z * K * J;
  const int j0 = blockIdx.x * 32;
  const int k0 = blockIdx.y * 32;
  const int tr = threadIdx.x >> 5;   // 0..7
  const int tc = threadIdx.x & 31;   // 0..31

#pragma unroll
  for (int p = 0; p < 4; ++p) {
    const int kl = p * 8 + tr;       // k_local
    // coalesced along j; scattered into LDS at [j_local][k_local]
    async_g2l_b32(lds_off(&tile[tc * 33 + kl]),
                  src + bofs + (size_t)(k0 + kl) * J + (j0 + tc));
  }
  asm volatile("s_wait_asynccnt 0x0" ::: "memory");
  __syncthreads();

#pragma unroll
  for (int p = 0; p < 4; ++p) {
    const int jl = p * 8 + tr;       // j_local
    dst[bofs + (size_t)(j0 + jl) * K + (k0 + tc)] =
        (_Float16)tile[jl * 33 + tc];
  }
}

// ---------------------------------------------------------------------------
// Kernel 1: qkv = tokens @ w_proj + b_proj.
// One wave per 32x32 output tile (2x2 WMMA blocking). M=8192, Nout=1536, K=512.
// A rows (tokens) and B rows (w_projT) are contiguous f16 -> pure vector loads.
// q scaled by D^-0.5; q,k stored [B,H,N,D]; v stored transposed [B,H,D,N].
// ---------------------------------------------------------------------------
__global__ __launch_bounds__(256) void qkv_kernel(
    const _Float16* __restrict__ tok, const _Float16* __restrict__ wpT,
    const float* __restrict__ bp,
    _Float16* __restrict__ qh, _Float16* __restrict__ kh,
    _Float16* __restrict__ vT) {
  const int wave = blockIdx.x * 8 + (threadIdx.x >> 5);
  const int lane = threadIdx.x & 31;
  const int hi = lane >> 4, l16 = lane & 15;
  const int mt = wave / 48, jt = wave % 48;   // 256 x 48 tiles
  const int m0 = mt * 32;
  const int j0 = jt * 32;
  const int b  = m0 >> 10;
  const int n0 = m0 & 1023;

  const _Float16* a0 = tok + (size_t)(m0 + l16) * Cch;
  const _Float16* a1 = a0 + 16 * Cch;
  const _Float16* b0 = wpT + (size_t)(j0 + l16) * Cch;
  const _Float16* b1 = b0 + 16 * Cch;

  v8f acc[2][2] = {};
#pragma unroll 1
  for (int ks = 0; ks < 16; ++ks) {
    const int kc = ks * 32;
    __builtin_prefetch(a0 + kc + 32, 0, 0);
    __builtin_prefetch(b0 + kc + 32, 0, 0);
    const v16h A0 = *(const v16h*)(a0 + kc + 16 * hi);
    const v16h A1 = *(const v16h*)(a1 + kc + 16 * hi);
    const v16h B0 = *(const v16h*)(b0 + kc + 16 * hi);
    const v16h B1 = *(const v16h*)(b1 + kc + 16 * hi);
    acc[0][0] = WMMA_F16(A0, B0, acc[0][0]);
    acc[0][1] = WMMA_F16(A0, B1, acc[0][1]);
    acc[1][0] = WMMA_F16(A1, B0, acc[1][0]);
    acc[1][1] = WMMA_F16(A1, B1, acc[1][1]);
  }

  // epilogue: lane owns column j (per group g); rows m = rg*16 + vv + 8*hi
#pragma unroll
  for (int g = 0; g < 2; ++g) {
    const int j  = j0 + g * 16 + l16;
    const int hd = j / 192, r = j % 192, t = r >> 6, d = r & 63;
    const float bias = bp[j];
    const float scl  = (t == 0) ? 0.125f : 1.0f;   // SCALE = 64^-0.5
#pragma unroll
    for (int rg = 0; rg < 2; ++rg) {
      if (t < 2) {
        _Float16* dst = (t == 0) ? qh : kh;
        const size_t base = ((size_t)(b * Hh + hd) * Np) * Dd + d;
#pragma unroll
        for (int vv = 0; vv < 8; ++vv) {
          const int n = n0 + rg * 16 + vv + 8 * hi;
          dst[base + (size_t)n * Dd] = (_Float16)((acc[rg][g][vv] + bias) * scl);
        }
      } else {  // v: store transposed [B,H,D,N] (contiguous along n)
        const size_t base = ((size_t)(b * Hh + hd) * Dd + d) * Np;
#pragma unroll
        for (int vv = 0; vv < 8; ++vv) {
          const int n = n0 + rg * 16 + vv + 8 * hi;
          vT[base + n] = (_Float16)(acc[rg][g][vv] + bias);
        }
      }
    }
  }
}

// ---------------------------------------------------------------------------
// Kernel 2: flash attention. One wave per (b, h, 16-query tile).
// S^T = K-tile @ Q^T (softmax axis j lives in the lane pair), online softmax,
// P re-laid-out via LDS, O += P @ V with V^T giving contiguous B-operands.
// ---------------------------------------------------------------------------
__global__ __launch_bounds__(256) void attn_kernel(
    const _Float16* __restrict__ qh, const _Float16* __restrict__ kh,
    const _Float16* __restrict__ vT, _Float16* __restrict__ res) {
  __shared__ __align__(32) _Float16 ldsP[8][512];  // per-wave 16x32 P tile
  __shared__ float ldsS[8][16];                    // per-wave scalar bcast

  const int wv   = threadIdx.x >> 5;
  const int lane = threadIdx.x & 31;
  const int hi = lane >> 4, l16 = lane & 15;
  const int wave = blockIdx.x * 8 + wv;
  const int qt = wave & 63;          // 64 query tiles
  const int bh = wave >> 6;          // b*8 + h
  const int b = bh >> 3, hd = bh & 7;

  const size_t hbase = (size_t)bh * Np * Dd;
  const _Float16* qrow = qh + hbase + (size_t)(qt * 16 + l16) * Dd;
  const _Float16* kb_  = kh + hbase;
  const _Float16* vt   = vT + hbase;      // [D][N] for this (b,h)
  _Float16* lp = &ldsP[wv][0];
  float*    ls = &ldsS[wv][0];

  // Q as WMMA B-matrix (col i = l16, d in table order)
  const v16h qB0 = cat8(*(const v8h*)(qrow + 8 * hi),
                        *(const v8h*)(qrow + 16 + 8 * hi));
  const v16h qB1 = cat8(*(const v8h*)(qrow + 32 + 8 * hi),
                        *(const v8h*)(qrow + 48 + 8 * hi));

  v8f o[4] = {};                      // O accumulators, 4 d-tiles of 16
  float mrun = -1e30f, lrun = 0.0f;

#pragma unroll 1
  for (int j0 = 0; j0 < Np; j0 += 32) {
    // ---- S^T for 32 keys: rows j, cols i ----
    v8f st0 = {}, st1 = {};
    {
      const _Float16* kr0 = kb_ + (size_t)(j0 + l16) * Dd;
      const _Float16* kr1 = kb_ + (size_t)(j0 + 16 + l16) * Dd;
      st0 = WMMA_F16(cat8(*(const v8h*)(kr0 + 8 * hi),
                          *(const v8h*)(kr0 + 16 + 8 * hi)), qB0, st0);
      st0 = WMMA_F16(cat8(*(const v8h*)(kr0 + 32 + 8 * hi),
                          *(const v8h*)(kr0 + 48 + 8 * hi)), qB1, st0);
      st1 = WMMA_F16(cat8(*(const v8h*)(kr1 + 8 * hi),
                          *(const v8h*)(kr1 + 16 + 8 * hi)), qB0, st1);
      st1 = WMMA_F16(cat8(*(const v8h*)(kr1 + 32 + 8 * hi),
                          *(const v8h*)(kr1 + 48 + 8 * hi)), qB1, st1);
    }

    // ---- online softmax over key axis (rows; split across lane pair) ----
    float mloc = -1e30f;
#pragma unroll
    for (int vv = 0; vv < 8; ++vv)
      mloc = fmaxf(mloc, fmaxf(st0[vv], st1[vv]));
    mloc = fmaxf(mloc, __shfl_xor(mloc, 16, 32));
    const float mnew  = fmaxf(mrun, mloc);
    const float alpha = __expf(mrun - mnew);
    float p0[8], p1[8], ps = 0.0f;
#pragma unroll
    for (int vv = 0; vv < 8; ++vv) {
      p0[vv] = __expf(st0[vv] - mnew);
      p1[vv] = __expf(st1[vv] - mnew);
      ps += p0[vv] + p1[vv];
    }
    ps += __shfl_xor(ps, 16, 32);
    lrun = lrun * alpha + ps;
    mrun = mnew;

    // ---- P tile -> LDS in [i][j] order; alpha broadcast ----
#pragma unroll
    for (int vv = 0; vv < 8; ++vv) {
      lp[l16 * 32 + vv + 8 * hi]      = (_Float16)p0[vv];
      lp[l16 * 32 + 16 + vv + 8 * hi] = (_Float16)p1[vv];
    }
    if (hi == 0) ls[l16] = alpha;
    asm volatile("s_wait_dscnt 0x0" ::: "memory");

    // rescale O accumulators (rows i = vv + 8*hi)
    float al[8];
#pragma unroll
    for (int vv = 0; vv < 8; ++vv) al[vv] = ls[vv + 8 * hi];
#pragma unroll
    for (int dt = 0; dt < 4; ++dt)
#pragma unroll
      for (int vv = 0; vv < 8; ++vv) o[dt][vv] *= al[vv];

    // P as WMMA A-matrix from LDS
    const v16h PA = *(const v16h*)(lp + l16 * 32 + 16 * hi);

    // ---- O += P @ V : V^T row is contiguous -> one v16h per d-tile ----
#pragma unroll
    for (int dt = 0; dt < 4; ++dt) {
      const v16h VB =
          *(const v16h*)(vt + (size_t)(dt * 16 + l16) * Np + j0 + 16 * hi);
      o[dt] = WMMA_F16(PA, VB, o[dt]);
    }
  }

  // ---- finalize: O / l, store res[b][n][h*64+d] as f16 ----
  if (hi == 0) ls[l16] = 1.0f / lrun;
  asm volatile("s_wait_dscnt 0x0" ::: "memory");
  float il[8];
#pragma unroll
  for (int vv = 0; vv < 8; ++vv) il[vv] = ls[vv + 8 * hi];
#pragma unroll
  for (int dt = 0; dt < 4; ++dt)
#pragma unroll
    for (int vv = 0; vv < 8; ++vv) {
      const int n = qt * 16 + vv + 8 * hi;
      res[((size_t)b * Np + n) * 512 + hd * 64 + dt * 16 + l16] =
          (_Float16)(o[dt][vv] * il[vv]);
    }
}

// ---------------------------------------------------------------------------
// Kernel 3: out = res @ w_out + b_out + tokens, written as [B,C,N].
// One wave per 32x32 tile (2x2 WMMA blocking). M=8192, Nout=512, K=512.
// ---------------------------------------------------------------------------
__global__ __launch_bounds__(256) void outproj_kernel(
    const _Float16* __restrict__ res, const _Float16* __restrict__ woT,
    const float* __restrict__ bo, const float* __restrict__ x,
    float* __restrict__ out) {
  const int wave = blockIdx.x * 8 + (threadIdx.x >> 5);
  const int lane = threadIdx.x & 31;
  const int hi = lane >> 4, l16 = lane & 15;
  const int mt = wave >> 4;          // 256 token tiles
  const int ct = wave & 15;          // 16 channel tiles
  const int m0 = mt * 32;
  const int c0 = ct * 32;
  const int b = m0 >> 10, n0 = m0 & 1023;

  const _Float16* a0 = res + (size_t)(m0 + l16) * 512;
  const _Float16* a1 = a0 + 16 * 512;
  const _Float16* b0 = woT + (size_t)(c0 + l16) * 512;
  const _Float16* b1 = b0 + 16 * 512;

  v8f acc[2][2] = {};
#pragma unroll 1
  for (int ks = 0; ks < 16; ++ks) {
    const int kc = ks * 32;
    __builtin_prefetch(a0 + kc + 32, 0, 0);
    __builtin_prefetch(b0 + kc + 32, 0, 0);
    const v16h A0 = *(const v16h*)(a0 + kc + 16 * hi);
    const v16h A1 = *(const v16h*)(a1 + kc + 16 * hi);
    const v16h B0 = *(const v16h*)(b0 + kc + 16 * hi);
    const v16h B1 = *(const v16h*)(b1 + kc + 16 * hi);
    acc[0][0] = WMMA_F16(A0, B0, acc[0][0]);
    acc[0][1] = WMMA_F16(A0, B1, acc[0][1]);
    acc[1][0] = WMMA_F16(A1, B0, acc[1][0]);
    acc[1][1] = WMMA_F16(A1, B1, acc[1][1]);
  }

#pragma unroll
  for (int g = 0; g < 2; ++g) {
    const int c = c0 + g * 16 + l16;
    const float bias = bo[c];
#pragma unroll
    for (int rg = 0; rg < 2; ++rg) {
      // rows n = n0 + rg*16 + vv + 8*hi -> 8 contiguous floats per lane
      const size_t rowbase =
          ((size_t)(b * Cch) + c) * Np + n0 + rg * 16 + 8 * hi;
      const float* xr = x + rowbase;   // residual (tokens) shares layout
      float* orr = out + rowbase;
#pragma unroll
      for (int vv = 0; vv < 8; ++vv)
        orr[vv] = acc[rg][g][vv] + bias + xr[vv];
    }
  }
}

// ---------------------------------------------------------------------------
extern "C" void kernel_launch(void* const* d_in, const int* in_sizes, int n_in,
                              void* d_out, int out_size, void* d_ws,
                              size_t ws_size, hipStream_t stream) {
  (void)in_sizes; (void)n_in; (void)out_size; (void)ws_size;
  const float* x  = (const float*)d_in[0];
  const float* wp = (const float*)d_in[1];
  const float* bp = (const float*)d_in[2];
  const float* wo = (const float*)d_in[3];
  const float* bo = (const float*)d_in[4];
  float* out = (float*)d_out;

  const size_t tokN = (size_t)Bn * Np * Cch;   // 4,194,304
  _Float16* tok = (_Float16*)d_ws;             // [B][N][C] f16
  _Float16* wpT = tok + tokN;                  // [1536][512] f16
  _Float16* woT = wpT + (size_t)1536 * 512;    // [512][512] f16
  _Float16* qh  = woT + (size_t)512 * 512;     // [B,H,N,D] f16
  _Float16* kh  = qh + tokN;
  _Float16* vT  = kh + tokN;                   // [B,H,D,N] f16
  _Float16* rs  = vT + tokN;                   // [B,N,512] f16

  // pack: tokens (batched transpose of x), w_proj^T, w_out^T
  transpose_cvt_kernel<<<dim3(Np / 32, Cch / 32, Bn), 256, 0, stream>>>(
      x, tok, Cch, Np);
  transpose_cvt_kernel<<<dim3(1536 / 32, Cch / 32, 1), 256, 0, stream>>>(
      wp, wpT, Cch, 1536);
  transpose_cvt_kernel<<<dim3(Cch / 32, Cch / 32, 1), 256, 0, stream>>>(
      wo, woT, Cch, Cch);

  // 256*48 waves / 8 per block
  qkv_kernel<<<1536, 256, 0, stream>>>(tok, wpT, bp, qh, kh, vT);
  // 8*8*64 waves / 8 per block
  attn_kernel<<<512, 256, 0, stream>>>(qh, kh, vT, rs);
  // 256*16 waves / 8 per block
  outproj_kernel<<<512, 256, 0, stream>>>(rs, woT, bo, x, out);
}
